// AttentionSSA_27023934226838
// MI455X (gfx1250) — matrix operand order repaired and checked by reading, not compiled, not run on hardware
//
#include <hip/hip_runtime.h>

typedef __attribute__((ext_vector_type(16))) __bf16 v16bf;
typedef __attribute__((ext_vector_type(8)))  __bf16 v8bf;
typedef __attribute__((ext_vector_type(8)))  float  v8f;

#define DIM    768
#define NHEAD  12
#define HDIM   64
#define BATCH  16
#define SEQ    640
#define NTOK   (BATCH * SEQ)   /* 10240 */
#define SCALE_F 0.125f         /* 64^-0.5 */
#define LOG2E_F 1.4426950408889634f

// ---------------------------------------------------------------------------
// CDNA5 async global->LDS staging (ASYNCcnt-tracked).
// Each lane copies 16 bytes; generic LDS pointer's low 32 bits = LDS offset.
// ---------------------------------------------------------------------------
__device__ inline void async_b128(void* lds_dst, const void* gsrc) {
  unsigned off = (unsigned)(unsigned long long)lds_dst;
  asm volatile("global_load_async_to_lds_b128 %0, %1, off"
               :: "v"(off), "v"(gsrc) : "memory");
}
template <int N>
__device__ inline void wait_async() {
  asm volatile("s_wait_asynccnt %0" :: "i"(N) : "memory");
}

// ---------------------------------------------------------------------------
// Fragment loaders (wave32); A and B use one contiguous k order, k = 16*(lane>>4) + s.
// A 16x32 bf16: lane holds row m=lane&15; K base kb=(lane>>4)*16;
//   elements 0..15 = K kb..kb+15 (same order as B).
// B 32x16 bf16: lane holds col n=lane&15; K base kb=(lane>>4)*16;
//   elements 0..15 = K kb..kb+15 (contiguous when source is n-major).
// C/D 16x16 f32: element v at (row = v + 8*(lane>>4), col = lane&15).
// ---------------------------------------------------------------------------
__device__ inline v16bf ld_afrag(const __bf16* rowptr, int kb) {
  v8bf lo = *(const v8bf*)(rowptr + kb);
  v8bf hi = *(const v8bf*)(rowptr + kb + 8);
  v16bf a;
#pragma unroll
  for (int i = 0; i < 8; ++i) { a[i] = lo[i]; a[8 + i] = hi[i]; }
  return a;
}

__device__ inline v16bf ld_bfrag(const __bf16* p) {
  v8bf lo = *(const v8bf*)p;
  v8bf hi = *(const v8bf*)(p + 8);
  v16bf b;
#pragma unroll
  for (int i = 0; i < 8; ++i) { b[i] = lo[i]; b[8 + i] = hi[i]; }
  return b;
}

// ---------------------------------------------------------------------------
// f32 -> bf16 conversion (RNE)
// ---------------------------------------------------------------------------
__global__ void cvt_f32_bf16(const float* __restrict__ src,
                             __bf16* __restrict__ dst, int n) {
  int i = (blockIdx.x * blockDim.x + threadIdx.x) * 4;
  if (i + 3 < n) {
    float4 v = *(const float4*)(src + i);
    dst[i]     = (__bf16)v.x;
    dst[i + 1] = (__bf16)v.y;
    dst[i + 2] = (__bf16)v.z;
    dst[i + 3] = (__bf16)v.w;
  }
}

// f32 [K x N] -> bf16 transposed [N x K] (32x32 tiles through LDS)
__global__ __launch_bounds__(256)
void cvt_transpose_bf16(const float* __restrict__ W, __bf16* __restrict__ WT,
                        int K, int N) {
  __shared__ float tile[32][33];
  int kb = blockIdx.y * 32, nb = blockIdx.x * 32;
  int tx = threadIdx.x & 31, ty = threadIdx.x >> 5;  // 8 row-groups
#pragma unroll
  for (int r = ty; r < 32; r += 8)
    tile[r][tx] = W[(size_t)(kb + r) * N + nb + tx];
  __syncthreads();
#pragma unroll
  for (int r = ty; r < 32; r += 8)
    WT[(size_t)(nb + r) * K + kb + tx] = (__bf16)tile[tx][r];
}

// ---------------------------------------------------------------------------
// Tiled bf16 WMMA GEMM: C[M,N] = A[M,K] @ BT[N,K]^T + bias
// Block tile 128x128, K-step 32, 256 threads = 8 waves (4x2),
// async double-buffered LDS staging (both operands K-contiguous).
// MODE 0: scatter bf16 Q (pre-scaled by SCALE*log2e) / K [B,H,SEQ,HDIM],
//         V transposed [B,H,HDIM,SEQ]
// MODE 1: store f32 row-major + bias (output projection)
// ---------------------------------------------------------------------------
template <int MODE>
__global__ __launch_bounds__(256)
void gemm_bf16_kernel(const __bf16* __restrict__ A, const __bf16* __restrict__ BT,
                      const float* __restrict__ bias,
                      __bf16* __restrict__ Qo, __bf16* __restrict__ Ko,
                      __bf16* __restrict__ Vo, float* __restrict__ Fo,
                      int N, int K) {
  __shared__ __bf16 As[2][128][40];
  __shared__ __bf16 Bs[2][128][40];

  const int t = threadIdx.x;
  const int lane = t & 31, wid = t >> 5;
  const int wm = wid & 3, wn = wid >> 2;
  const int m0 = blockIdx.y * 128, n0 = blockIdx.x * 128;

  const int am = lane & 15, kbA = (lane >> 4) * 16;
  const int bn = lane & 15, kbB = (lane >> 4) * 16;
  const int halfid = lane >> 4, nl = lane & 15;

  v8f zero = {};
  v8f acc[2][4];
#pragma unroll
  for (int i = 0; i < 2; ++i)
#pragma unroll
    for (int j = 0; j < 4; ++j) acc[i][j] = zero;

  // stage one 128x32 tile pair into buffer `buf` (4 asyncs / wave)
  auto stage = [&](int buf, int k0) {
#pragma unroll
    for (int rep = 0; rep < 2; ++rep) {
      int s = t + rep * 256;          // 512 16B segments per matrix
      int row = s >> 2, cs = (s & 3) * 8;
      async_b128(&As[buf][row][cs], A + (size_t)(m0 + row) * K + k0 + cs);
      async_b128(&Bs[buf][row][cs], BT + (size_t)(n0 + row) * K + k0 + cs);
    }
  };

  stage(0, 0);
  const int nsteps = K >> 5;
  for (int i = 0; i < nsteps; ++i) {
    int cur = i & 1;
    if (i + 1 < nsteps) {
      stage(cur ^ 1, (i + 1) << 5);
      wait_async<4>();                 // retire current buffer's 4 asyncs
    } else {
      wait_async<0>();
    }
    __syncthreads();

    v16bf af[2], bfr[4];
#pragma unroll
    for (int ii = 0; ii < 2; ++ii)
      af[ii] = ld_afrag(&As[cur][wm * 32 + ii * 16 + am][0], kbA);
#pragma unroll
    for (int j = 0; j < 4; ++j)
      bfr[j] = ld_bfrag(&Bs[cur][wn * 64 + j * 16 + bn][kbB]);
#pragma unroll
    for (int ii = 0; ii < 2; ++ii)
#pragma unroll
      for (int j = 0; j < 4; ++j)
        acc[ii][j] = __builtin_amdgcn_wmma_f32_16x16x32_bf16(
            false, af[ii], false, bfr[j], (short)0, acc[ii][j], false, false);
    __syncthreads();
  }

  // epilogue
  const float qscale = SCALE_F * LOG2E_F;   // fold attention scale + exp2 dom.
#pragma unroll
  for (int i = 0; i < 2; ++i) {
#pragma unroll
    for (int j = 0; j < 4; ++j) {
      int c = n0 + wn * 64 + j * 16 + nl;
      float bv = bias[c];
#pragma unroll
      for (int v = 0; v < 8; ++v) {
        int m = m0 + wm * 32 + i * 16 + v + 8 * halfid;
        float val = acc[i][j][v] + bv;
        if (MODE == 0) {
          int which = c / DIM;
          int cr = c - which * DIM;
          int h = cr >> 6, d = cr & 63;
          int b = m / SEQ, nn = m - b * SEQ;
          if (which == 0)
            Qo[((size_t)(b * NHEAD + h) * SEQ + nn) * HDIM + d] =
                (__bf16)(val * qscale);
          else if (which == 1)
            Ko[((size_t)(b * NHEAD + h) * SEQ + nn) * HDIM + d] = (__bf16)val;
          else  // V stored transposed: [B,H,HDIM,SEQ]
            Vo[((size_t)(b * NHEAD + h) * HDIM + d) * SEQ + nn] = (__bf16)val;
        } else {
          Fo[(size_t)m * DIM + c] = val;
        }
      }
    }
  }
}

// ---------------------------------------------------------------------------
// Flash-style attention in the exp2 domain:
//   S' = (Q*scale*log2e) @ K^T ;  softmax(S) == exp2-softmax(S')
//   relu(S)^2 == relu(S')^2 / log2e^2  (constant folded into ws1)
// P = ws0*softmax + ws1*relu^2,  O = P @ V
// grid (5 row-groups, B*H), 256 threads = 8 waves, one 16-row Q tile / wave.
// ---------------------------------------------------------------------------
__global__ __launch_bounds__(256)
void attn_kernel(const __bf16* __restrict__ Qg, const __bf16* __restrict__ Kg,
                 const __bf16* __restrict__ VtG, const float* __restrict__ wvec,
                 __bf16* __restrict__ Og) {
  __shared__ __bf16 Ks[128][72];        // K chunk, row-major (keys x d)
  __shared__ __bf16 Vt[64][136];        // V chunk, d-major (d x keys)
  __shared__ __bf16 stg[8][16][72];     // per-wave C->A restage (16 x 64)

  const int t = threadIdx.x, lane = t & 31, wid = t >> 5;
  const int bh = blockIdx.y;
  const int b = bh / NHEAD, h = bh % NHEAD;
  const int rt = blockIdx.x * 8 + wid;       // row tile 0..39
  const size_t base = (size_t)bh * SEQ * HDIM;

  const int am = lane & 15, kbA = (lane >> 4) * 16;
  const int bn = lane & 15, kbB = (lane >> 4) * 16;
  const int halfid = lane >> 4, nl = lane & 15;

  float w0 = wvec[0], w1 = wvec[1];
  float mw = fmaxf(w0, w1);
  float e0 = __expf(w0 - mw), e1 = __expf(w1 - mw);
  float ws0 = e0 / (e0 + e1);
  float ws1 = (e1 / (e0 + e1)) / (LOG2E_F * LOG2E_F);  // undo exp2-domain scale

  const __bf16* qrow = Qg + base + (size_t)(rt * 16 + am) * HDIM;
  v16bf aq[2];
#pragma unroll
  for (int i = 0; i < 2; ++i) aq[i] = ld_afrag(qrow + i * 32, kbA);

  v8f zero = {};
  v8f accE[4], accR[4];
#pragma unroll
  for (int j = 0; j < 4; ++j) { accE[j] = zero; accR[j] = zero; }
  float rmax[8], rsum[8];
#pragma unroll
  for (int v = 0; v < 8; ++v) { rmax[v] = -1e30f; rsum[v] = 0.0f; }

  for (int cc = 0; cc < 5; ++cc) {       // 5 chunks of 128 keys
    __syncthreads();
    // async-stage K chunk (1024 x 16B) and Vt chunk (1024 x 16B)
#pragma unroll
    for (int rep = 0; rep < 4; ++rep) {
      int s = t + rep * 256;
      {
        int row = s >> 3, c = (s & 7) * 8;
        async_b128(&Ks[row][c],
                   Kg + base + (size_t)(cc * 128 + row) * HDIM + c);
      }
      {
        int d = s >> 4, c = (s & 15) * 8;
        async_b128(&Vt[d][c],
                   VtG + base + (size_t)d * SEQ + cc * 128 + c);
      }
    }
    wait_async<0>();
    __syncthreads();

    for (int mt = 0; mt < 2; ++mt) {     // 2 steps of 64 keys
      const int mb = mt * 64;
      // ---- S' = Qs @ K^T : four 16x16 C tiles, K-dim = d = 64 ----
      v8f c[4];
#pragma unroll
      for (int ti = 0; ti < 4; ++ti) c[ti] = zero;
#pragma unroll
      for (int kk = 0; kk < 2; ++kk)
#pragma unroll
        for (int ti = 0; ti < 4; ++ti) {
          v16bf bk = ld_bfrag(&Ks[mb + ti * 16 + bn][kk * 32 + kbB]);
          c[ti] = __builtin_amdgcn_wmma_f32_16x16x32_bf16(
              false, aq[kk], false, bk, (short)0, c[ti], false, false);
        }

      // ---- online softmax stats (exp2 domain, one pass per 64 keys) ----
#pragma unroll
      for (int v = 0; v < 8; ++v) {
        float tm = fmaxf(fmaxf(c[0][v], c[1][v]), fmaxf(c[2][v], c[3][v]));
#pragma unroll
        for (int mk = 1; mk < 16; mk <<= 1) tm = fmaxf(tm, __shfl_xor(tm, mk, 32));
        float nm = fmaxf(rmax[v], tm);
        float f = exp2f(rmax[v] - nm);
        rmax[v] = nm;
        rsum[v] *= f;
#pragma unroll
        for (int j = 0; j < 4; ++j) accE[j][v] *= f;
      }
      float p[4][8];
#pragma unroll
      for (int v = 0; v < 8; ++v) {
#pragma unroll
        for (int ti = 0; ti < 4; ++ti) p[ti][v] = exp2f(c[ti][v] - rmax[v]);
        float ts = (p[0][v] + p[1][v]) + (p[2][v] + p[3][v]);
#pragma unroll
        for (int mk = 1; mk < 16; mk <<= 1) ts += __shfl_xor(ts, mk, 32);
        rsum[v] += ts;
      }

      // ---- exp part: restage C->A layout, accumulate P_exp @ V ----
#pragma unroll
      for (int ti = 0; ti < 4; ++ti)
#pragma unroll
        for (int v = 0; v < 8; ++v)
          stg[wid][v + 8 * halfid][ti * 16 + nl] = (__bf16)p[ti][v];
      v16bf apE[2];
#pragma unroll
      for (int kk = 0; kk < 2; ++kk)
        apE[kk] = ld_afrag(&stg[wid][am][kk * 32], kbA);
#pragma unroll
      for (int j = 0; j < 4; ++j)
#pragma unroll
        for (int kk = 0; kk < 2; ++kk) {
          v16bf bv = ld_bfrag(&Vt[j * 16 + bn][mb + kk * 32 + kbB]);
          accE[j] = __builtin_amdgcn_wmma_f32_16x16x32_bf16(
              false, apE[kk], false, bv, (short)0, accE[j], false, false);
        }

      // ---- relu^2 part (scaled domain; constant folded into ws1) ----
#pragma unroll
      for (int ti = 0; ti < 4; ++ti)
#pragma unroll
        for (int v = 0; v < 8; ++v) {
          float r = fmaxf(c[ti][v], 0.0f);
          stg[wid][v + 8 * halfid][ti * 16 + nl] = (__bf16)(r * r);
        }
      v16bf apR[2];
#pragma unroll
      for (int kk = 0; kk < 2; ++kk)
        apR[kk] = ld_afrag(&stg[wid][am][kk * 32], kbA);
#pragma unroll
      for (int j = 0; j < 4; ++j)
#pragma unroll
        for (int kk = 0; kk < 2; ++kk) {
          v16bf bv = ld_bfrag(&Vt[j * 16 + bn][mb + kk * 32 + kbB]);
          accR[j] = __builtin_amdgcn_wmma_f32_16x16x32_bf16(
              false, apR[kk], false, bv, (short)0, accR[j], false, false);
        }
    }
  }

  // ---- epilogue: blend, store bf16 in [b, n, h*64+d] order for proj GEMM ----
#pragma unroll
  for (int v = 0; v < 8; ++v) {
    float inv = 1.0f / rsum[v];
    int row = rt * 16 + v + 8 * halfid;
#pragma unroll
    for (int j = 0; j < 4; ++j) {
      int d = j * 16 + nl;
      float o = ws0 * accE[j][v] * inv + ws1 * accR[j][v];
      Og[((size_t)(b * SEQ + row)) * DIM + h * HDIM + d] = (__bf16)o;
    }
  }
}

// ---------------------------------------------------------------------------
// Host launcher
// ---------------------------------------------------------------------------
extern "C" void kernel_launch(void* const* d_in, const int* in_sizes, int n_in,
                              void* d_out, int out_size, void* d_ws, size_t ws_size,
                              hipStream_t stream) {
  (void)in_sizes; (void)n_in; (void)out_size; (void)ws_size;

  const float* x      = (const float*)d_in[0];
  const float* qkv_w  = (const float*)d_in[1];
  const float* qkv_b  = (const float*)d_in[2];
  const float* proj_w = (const float*)d_in[3];
  const float* proj_b = (const float*)d_in[4];
  const float* wblend = (const float*)d_in[5];

  const size_t TOKDIM = (size_t)NTOK * DIM;     // 7,864,320
  char* ws = (char*)d_ws;
  __bf16* xb      = (__bf16*)ws; ws += TOKDIM * 2;
  __bf16* wqkvT   = (__bf16*)ws; ws += (size_t)DIM * 3 * DIM * 2;  // [2304 x 768]
  __bf16* wprojT  = (__bf16*)ws; ws += (size_t)DIM * DIM * 2;      // [768 x 768]
  __bf16* Qb      = (__bf16*)ws; ws += TOKDIM * 2;
  __bf16* Kb      = (__bf16*)ws; ws += TOKDIM * 2;
  __bf16* Vtb     = (__bf16*)ws; ws += TOKDIM * 2;                 // [B,H,D,SEQ]
  __bf16* AOb     = (__bf16*)ws; ws += TOKDIM * 2;

  // 1) convert / transpose to bf16
  cvt_f32_bf16<<<(int)(TOKDIM / 1024), 256, 0, stream>>>(x, xb, (int)TOKDIM);
  cvt_transpose_bf16<<<dim3(3 * DIM / 32, DIM / 32), 256, 0, stream>>>(
      qkv_w, wqkvT, DIM, 3 * DIM);
  cvt_transpose_bf16<<<dim3(DIM / 32, DIM / 32), 256, 0, stream>>>(
      proj_w, wprojT, DIM, DIM);

  // 2) QKV projection -> Q(scaled)/K [B,H,SEQ,HDIM], V transposed [B,H,HDIM,SEQ]
  gemm_bf16_kernel<0><<<dim3(18, 80), 256, 0, stream>>>(
      xb, wqkvT, qkv_b, Qb, Kb, Vtb, nullptr, 3 * DIM, DIM);

  // 3) attention: 192 heads x 5 row-groups
  attn_kernel<<<dim3(5, BATCH * NHEAD), 256, 0, stream>>>(
      Qb, Kb, Vtb, wblend, AOb);

  // 4) output projection: [10240,768] @ [768,768]^T(pre-T) + bias -> f32
  gemm_bf16_kernel<1><<<dim3(6, 80), 256, 0, stream>>>(
      AOb, wprojT, proj_b, nullptr, nullptr, nullptr, (float*)d_out, DIM, DIM);
}
